// GATLayer_64604898066674
// MI455X (gfx1250) — matrix operand order repaired and checked by reading, not compiled, not run on hardware
//
#include <hip/hip_runtime.h>
#include <math.h>

typedef __attribute__((ext_vector_type(16))) __bf16 v16bf;
typedef __attribute__((ext_vector_type(8)))  __bf16 v8bf;
typedef __attribute__((ext_vector_type(8)))  float  v8f;

#define SEQ_  1024
#define BSZ_  16
#define D_    512
#define H_    8
#define C0_   64
#define NNODE (SEQ_ * BSZ_)

static __device__ __forceinline__ float leaky02(float v) { return v > 0.f ? v : 0.2f * v; }
static __device__ __forceinline__ float eluf(float v)    { return v > 0.f ? v : expm1f(v); }

static __device__ __forceinline__ void atomicMaxF(float* addr, float val) {
  if (val >= 0.f) atomicMax((int*)addr, __float_as_int(val));
  else            atomicMin((unsigned int*)addr, __float_as_uint(val));
}

// x[s][b][d] -> nodes_bf16[(b*SEQ+s)][d]
__global__ void k_x_to_nodes_bf(const float* __restrict__ x, __bf16* __restrict__ nb) {
  long long t = (long long)blockIdx.x * blockDim.x + threadIdx.x;
  const long long total = (long long)SEQ_ * BSZ_ * D_;
  if (t >= total) return;
  int dcol = (int)(t % D_);
  long long sb = t / D_;
  int b = (int)(sb % BSZ_);
  int s = (int)(sb / BSZ_);
  nb[((size_t)b * SEQ_ + s) * (size_t)D_ + dcol] = (__bf16)x[t];
}

// W[k][n] (row-major KxN) -> Wt[n][k] bf16 (so B fragments are contiguous per lane)
__global__ void k_packWt(const float* __restrict__ W, __bf16* __restrict__ Wt,
                         int K, int Ncols) {
  long long t = (long long)blockIdx.x * blockDim.x + threadIdx.x;
  if (t >= (long long)K * Ncols) return;
  int k = (int)(t % K);
  long long n = t / K;
  Wt[t] = (__bf16)W[(size_t)k * Ncols + n];
}

// C[M=16384, N=512] f32 = A_bf16[M,512] * Wt_bf16[512 cols][512 K] (Wt row n = column n of W)
// 8 waves/block; wave -> 16x64 strip. grid = (2, 512).
__global__ __launch_bounds__(256) void k_gemm_bf16(const __bf16* __restrict__ A,
                                                   const __bf16* __restrict__ Bt,
                                                   float* __restrict__ C) {
  const int K = 512, NC = 512;
  const int lane = threadIdx.x & 31;
  const int w    = threadIdx.x >> 5;
  const int wm   = w & 1;        // 2 wave rows  (32 M per block)
  const int wn   = w >> 1;       // 4 wave cols  (256 N per block)
  const int half = lane >> 4;
  const int ll   = lane & 15;
  const int mrow  = blockIdx.y * 32 + wm * 16 + ll;
  const int nbase = blockIdx.x * 256 + wn * 64;

  v8f acc[4];
#pragma unroll
  for (int t = 0; t < 4; ++t)
#pragma unroll
    for (int r = 0; r < 8; ++r) acc[t][r] = 0.f;

  const __bf16* arow = A + (size_t)mrow * K;
  for (int k0 = 0; k0 < K; k0 += 32) {
    // A 16x32 bf16 fragment: elems 0..15 -> K = k0+16*half+0..15 (contiguous)
    v8bf a0 = *(const v8bf*)(arow + k0 + 16 * half);
    v8bf a1 = *(const v8bf*)(arow + k0 + 16 * half + 8);
    v16bf av = __builtin_shufflevector(a0, a1, 0,1,2,3,4,5,6,7,8,9,10,11,12,13,14,15);
#pragma unroll
    for (int t = 0; t < 4; ++t) {
      // B 32x16 fragment: lane = col (ll), K = k0 + 16*half + 0..15 (contiguous in Wt)
      const __bf16* bcol = Bt + (size_t)(nbase + t * 16 + ll) * K + k0 + 16 * half;
      v8bf b0 = *(const v8bf*)bcol;
      v8bf b1 = *(const v8bf*)(bcol + 8);
      v16bf bv = __builtin_shufflevector(b0, b1, 0,1,2,3,4,5,6,7,8,9,10,11,12,13,14,15);
      acc[t] = __builtin_amdgcn_wmma_f32_16x16x32_bf16(
          /*neg_a=*/false, av, /*neg_b=*/false, bv,
          /*c_mod=*/(short)0, acc[t], /*reuse_a=*/false, /*reuse_b=*/false);
    }
  }
  // C fragment: VGPR r -> M = mbase + 8*half + r, N = nbase + 16*t + ll
  const int mb = blockIdx.y * 32 + wm * 16 + 8 * half;
#pragma unroll
  for (int t = 0; t < 4; ++t)
#pragma unroll
    for (int r = 0; r < 8; ++r)
      C[(size_t)(mb + r) * NC + nbase + t * 16 + ll] = acc[t][r];
}

// warp per (node, head): alpha_src/dst dot products
__global__ void k_alpha(const float* __restrict__ Hm, const float* __restrict__ att_s,
                        const float* __restrict__ att_d, float* __restrict__ as,
                        float* __restrict__ ad, int Nn, int Hh, int C) {
  long long warp = ((long long)blockIdx.x * blockDim.x + threadIdx.x) >> 5;
  int lane = threadIdx.x & 31;
  if (warp >= (long long)Nn * Hh) return;
  int n = (int)(warp / Hh), h = (int)(warp % Hh);
  const float* row = Hm + (size_t)n * Hh * C + (size_t)h * C;
  const float* vs  = att_s + (size_t)h * C;
  const float* vd  = att_d + (size_t)h * C;
  float ss = 0.f, sd = 0.f;
  for (int c = lane; c < C; c += 32) { float hv = row[c]; ss += hv * vs[c]; sd += hv * vd[c]; }
  for (int off = 16; off; off >>= 1) { ss += __shfl_xor(ss, off, 32); sd += __shfl_xor(sd, off, 32); }
  if (lane == 0) { as[warp] = ss; ad[warp] = sd; }
}

__global__ void k_fill(float* __restrict__ p, long long n, float v) {
  long long t = (long long)blockIdx.x * blockDim.x + threadIdx.x;
  if (t < n) p[t] = v;
}

__global__ void k_m_fixup(float* __restrict__ m, long long n) {
  long long t = (long long)blockIdx.x * blockDim.x + threadIdx.x;
  if (t < n) { float v = m[t]; if (!__builtin_isfinite(v)) m[t] = 0.f; }
}

__global__ void k_edge_max(const int* __restrict__ srcA, const int* __restrict__ dstA,
                           long long Eext, long long Eedges, int Hh,
                           const float* __restrict__ as, const float* __restrict__ ad,
                           float* __restrict__ m) {
  long long t = (long long)blockIdx.x * blockDim.x + threadIdx.x;
  if (t >= Eext * Hh) return;
  long long e = t / Hh; int h = (int)(t % Hh);
  int s, d;
  if (e < Eedges) { s = srcA[e]; d = dstA[e]; } else { s = d = (int)(e - Eedges); }
  float ev = leaky02(as[(size_t)s * Hh + h] + ad[(size_t)d * Hh + h]);
  atomicMaxF(&m[(size_t)d * Hh + h], ev);
}

__global__ void k_edge_expsum(const int* __restrict__ srcA, const int* __restrict__ dstA,
                              long long Eext, long long Eedges, int Hh,
                              const float* __restrict__ as, const float* __restrict__ ad,
                              const float* __restrict__ m, float* __restrict__ z) {
  long long t = (long long)blockIdx.x * blockDim.x + threadIdx.x;
  if (t >= Eext * Hh) return;
  long long e = t / Hh; int h = (int)(t % Hh);
  int s, d;
  if (e < Eedges) { s = srcA[e]; d = dstA[e]; } else { s = d = (int)(e - Eedges); }
  float ev = leaky02(as[(size_t)s * Hh + h] + ad[(size_t)d * Hh + h]);
  float p = expf(ev - m[(size_t)d * Hh + h]);
  atomicAdd(&z[(size_t)d * Hh + h], p);
}

// warp per (edge, head): out[dst, h*C + c] += h[src, h*C + c] * alpha
__global__ void k_edge_scatter(const int* __restrict__ srcA, const int* __restrict__ dstA,
                               long long Eext, long long Eedges, int Hh, int C,
                               const float* __restrict__ as, const float* __restrict__ ad,
                               const float* __restrict__ m, const float* __restrict__ z,
                               const float* __restrict__ Hsrc, int ldH,
                               float* __restrict__ out, int ldO) {
  long long warp = ((long long)blockIdx.x * blockDim.x + threadIdx.x) >> 5;
  int lane = threadIdx.x & 31;
  if (warp >= Eext * Hh) return;
  long long e = warp / Hh; int h = (int)(warp % Hh);
  int s, d;
  if (e < Eedges) { s = srcA[e]; d = dstA[e]; } else { s = d = (int)(e - Eedges); }
  float ev = leaky02(as[(size_t)s * Hh + h] + ad[(size_t)d * Hh + h]);
  float p  = expf(ev - m[(size_t)d * Hh + h]);
  float alpha = p / (z[(size_t)d * Hh + h] + 1e-16f);
  const float* hr = Hsrc + (size_t)s * ldH + (size_t)h * C;
  float* orow     = out  + (size_t)d * ldO + (size_t)h * C;
  for (int c = lane; c < C; c += 32) atomicAdd(&orow[c], hr[c] * alpha);
}

// v = elu(acc*scale + bias[c]); optional f32 out, optional bf16 out
__global__ void k_bias_elu(const float* __restrict__ accin, const float* __restrict__ bias,
                           float scale, float* __restrict__ outF, __bf16* __restrict__ outBf,
                           long long total, int Dd) {
  long long t = (long long)blockIdx.x * blockDim.x + threadIdx.x;
  if (t >= total) return;
  int c = (int)(t % Dd);
  float v = eluf(accin[t] * scale + bias[c]);
  if (outF)  outF[t]  = v;
  if (outBf) outBf[t] = (__bf16)v;
}

extern "C" void kernel_launch(void* const* d_in, const int* in_sizes, int n_in,
                              void* d_out, int out_size, void* d_ws, size_t ws_size,
                              hipStream_t stream) {
  const float* x    = (const float*)d_in[0];
  const int*   ei   = (const int*)  d_in[1];
  const float* W0   = (const float*)d_in[2];
  const float* as0w = (const float*)d_in[3];   // [8,64]
  const float* ad0w = (const float*)d_in[4];
  const float* b0   = (const float*)d_in[5];
  const float* W1   = (const float*)d_in[6];   // [512,4096]
  const float* as1w = (const float*)d_in[7];   // [8,512]
  const float* ad1w = (const float*)d_in[8];
  const float* b1   = (const float*)d_in[9];

  const long long Eedges = (long long)in_sizes[1] / 2;
  const int* srcA = ei;
  const int* dstA = ei + Eedges;
  const long long Eext = Eedges + NNODE;
  const long long ND   = (long long)NNODE * D_;

  // ---- workspace carve ----
  char* ws = (char*)d_ws; size_t off = 0;
  auto carve = [&](size_t bytes) { void* p = ws + off; off += (bytes + 255) & ~(size_t)255; return p; };
  __bf16* nodes_bf = (__bf16*)carve(ND * 2);                 // 16 MB
  __bf16* wt0      = (__bf16*)carve((size_t)D_ * D_ * 2);    // 0.5 MB
  __bf16* wt1      = (__bf16*)carve((size_t)D_ * H_ * D_ * 2); // 4 MB
  float*  hbuf     = (float*)carve(ND * 4);                  // 32 MB  (h0, then h1 per head)
  float*  as       = (float*)carve((size_t)NNODE * H_ * 4);
  float*  ad       = (float*)carve((size_t)NNODE * H_ * 4);
  float*  mbuf     = (float*)carve((size_t)NNODE * H_ * 4);
  float*  zbuf     = (float*)carve((size_t)NNODE * H_ * 4);
  float*  accbuf   = (float*)carve(ND * 4);                  // 32 MB  (out0, then head-sum acc)
  __bf16* x1bf     = (__bf16*)carve(ND * 2);                 // 16 MB
  (void)ws_size; (void)n_in; (void)out_size;

  const int TB = 256;
  auto nb = [&](long long work) { return (int)((work + TB - 1) / TB); };

  // ---- prep ----
  k_x_to_nodes_bf<<<nb(ND), TB, 0, stream>>>(x, nodes_bf);
  k_packWt<<<nb((long long)D_ * D_), TB, 0, stream>>>(W0, wt0, D_, D_);
  k_packWt<<<nb((long long)D_ * H_ * D_), TB, 0, stream>>>(W1, wt1, D_, H_ * D_);

  // ================= layer 0 (H=8, C=64, fused heads) =================
  k_gemm_bf16<<<dim3(2, NNODE / 32), TB, 0, stream>>>(nodes_bf, wt0, hbuf);
  k_alpha<<<nb((long long)NNODE * H_ * 32), TB, 0, stream>>>(hbuf, as0w, ad0w, as, ad, NNODE, H_, C0_);
  k_fill<<<nb((long long)NNODE * H_), TB, 0, stream>>>(mbuf, (long long)NNODE * H_, -__builtin_inff());
  hipMemsetAsync(zbuf, 0, (size_t)NNODE * H_ * 4, stream);
  hipMemsetAsync(accbuf, 0, (size_t)ND * 4, stream);
  k_edge_max<<<nb(Eext * H_), TB, 0, stream>>>(srcA, dstA, Eext, Eedges, H_, as, ad, mbuf);
  k_m_fixup<<<nb((long long)NNODE * H_), TB, 0, stream>>>(mbuf, (long long)NNODE * H_);
  k_edge_expsum<<<nb(Eext * H_), TB, 0, stream>>>(srcA, dstA, Eext, Eedges, H_, as, ad, mbuf, zbuf);
  k_edge_scatter<<<nb(Eext * H_ * 32), TB, 0, stream>>>(srcA, dstA, Eext, Eedges, H_, C0_,
                                                        as, ad, mbuf, zbuf, hbuf, D_, accbuf, D_);
  // x1 = elu(out0 + b0)  (bf16 copy only; f32 not needed downstream)
  k_bias_elu<<<nb(ND), TB, 0, stream>>>(accbuf, b0, 1.0f, (float*)nullptr, x1bf, ND, D_);

  // ================= layer 1 (per head; C=512; mean over heads) =================
  hipMemsetAsync(accbuf, 0, (size_t)ND * 4, stream);   // becomes head-sum accumulator
  for (int hh = 0; hh < H_; ++hh) {
    k_gemm_bf16<<<dim3(2, NNODE / 32), TB, 0, stream>>>(x1bf, wt1 + (size_t)hh * D_ * D_, hbuf);
    k_alpha<<<nb((long long)NNODE * 32), TB, 0, stream>>>(hbuf, as1w + (size_t)hh * D_,
                                                          ad1w + (size_t)hh * D_, as, ad, NNODE, 1, D_);
    k_fill<<<nb((long long)NNODE), TB, 0, stream>>>(mbuf, (long long)NNODE, -__builtin_inff());
    hipMemsetAsync(zbuf, 0, (size_t)NNODE * 4, stream);
    k_edge_max<<<nb(Eext), TB, 0, stream>>>(srcA, dstA, Eext, Eedges, 1, as, ad, mbuf);
    k_m_fixup<<<nb((long long)NNODE), TB, 0, stream>>>(mbuf, (long long)NNODE);
    k_edge_expsum<<<nb(Eext), TB, 0, stream>>>(srcA, dstA, Eext, Eedges, 1, as, ad, mbuf, zbuf);
    k_edge_scatter<<<nb(Eext * 32), TB, 0, stream>>>(srcA, dstA, Eext, Eedges, 1, D_,
                                                     as, ad, mbuf, zbuf, hbuf, D_, accbuf, D_);
  }
  // out = elu(acc/8 + b1); raw-reinterpret view == flat h, so write straight to d_out
  k_bias_elu<<<nb(ND), TB, 0, stream>>>(accbuf, b1, 1.0f / H_, (float*)d_out, (__bf16*)nullptr, ND, D_);
}
